// MatchModel_23003844837506
// MI455X (gfx1250) — matrix operand order repaired and checked by reading, not compiled, not run on hardware
//
#include <hip/hip_runtime.h>
#include <hip/hip_bf16.h>

// ---------------------------------------------------------------------------
// MI455X (gfx1250) implementation of the pick/place/match conv-stack model.
// Conv layers with Cout=256 run as implicit GEMM on v_wmma_f32_16x16x32_bf16.
// Block tile: 128(M=Cout) x 64(N = 2 rows x 32 cols); 8 waves, 4 WMMA accs
// per wave reusing one A (weight) fragment. Per 32-channel K-slice the full
// 4-row x 48-col input halo window is staged to LDS ONCE and reused by all
// 9 taps (36 WMMAs per wave per stage).
// ---------------------------------------------------------------------------

typedef __bf16 bf16;
typedef __attribute__((ext_vector_type(16))) __bf16 v16bf;
typedef __attribute__((ext_vector_type(8)))  __bf16 v8bf;
typedef __attribute__((ext_vector_type(8)))  float  v8f;

#define CSTR 40          // bf16 per LDS cell (32 ch + pad): 80B, 16B aligned, bank-spread
#define LROWS 4          // staged input rows: h0-1 .. h0+2
#define LCOLS 48         // staged input cols: w0-8 .. w0+39 (aligned 8-col runs)

// ---------------------------------------------------------------------------
// f32 -> bf16 bulk convert
// ---------------------------------------------------------------------------
__global__ void k_f32_to_bf16(const float* __restrict__ a, bf16* __restrict__ b, long n) {
    long i = (long)blockIdx.x * blockDim.x + threadIdx.x;
    if (i < n) b[i] = (bf16)a[i];
}

// Repack conv weight [O][I][3][3] f32 -> [tap=kh*3+kw][O][I] bf16
__global__ void k_repack_w(const float* __restrict__ w, bf16* __restrict__ o, int O, int I) {
    long idx = (long)blockIdx.x * blockDim.x + threadIdx.x;
    long total = (long)O * I * 9;
    if (idx >= total) return;
    int ci = (int)(idx % I); long t = idx / I;
    int co = (int)(t % O);   int tap = (int)(t / O);
    o[idx] = (bf16)w[((long)(co * I + ci)) * 9 + tap];
}

// ---------------------------------------------------------------------------
// 3x3 SAME conv, bf16 in/out, f32 WMMA accumulation.
// GEMM: M=Cout, N=64 positions (output rows h0,h0+1 x 32 cols), K=9*Cin.
// Block = 256 threads = 8 waves; wave w owns M rows [16w,16w+16) with four
// 16x16 accumulators: (row h0, n0..15), (row h0, n16..31), (row h0+1, ...).
// ---------------------------------------------------------------------------
__global__ __launch_bounds__(256)
void k_conv3x3_wmma(const bf16* __restrict__ in, const bf16* __restrict__ wt,
                    const float* __restrict__ bias, bf16* __restrict__ out,
                    int Cin, int Cout, int H, int W, int relu)
{
    __shared__ bf16 lds[LROWS * LCOLS * CSTR];   // 15,360 B

    const int tiles_w = W >> 5;
    int bt = blockIdx.x;
    const int w0 = (bt % tiles_w) * 32; bt /= tiles_w;
    const int h0 = (bt % (H >> 1)) * 2;          // output row pair (H is even)
    const int b  = bt / (H >> 1);
    const int coBase = blockIdx.y * 128;

    const int tid  = threadIdx.x;
    const int wave = tid >> 5;
    const int lane = tid & 31;
    const int l16  = lane & 15;
    const int halfSel = lane >> 4;          // ISA 7.12.2: lane half selects K range
    const int coWave  = coBase + wave * 16;
    const int kOfs = halfSel * 8;

    v8f acc[4] = {v8f{}, v8f{}, v8f{}, v8f{}};

    const bf16* inB = in + (long)b * Cin * H * W;
    const long chStride = (long)H * W;

    // ---- staging decode (kb-invariant): 32ch x 4rows x 6 runs of 8 cols ----
    // combo = tid + 256*k, k=0..2 : run = combo%6, row = (combo/6)%4, ch = combo/24
    int  sRun[3], sRow[3], sCh[3];
    bool sOK[3];
    long sOff[3];
    #pragma unroll
    for (int k = 0; k < 3; ++k) {
        int combo = tid + 256 * k;
        int run = combo % 6;
        int row = (combo / 6) % 4;
        int ch  = combo / 24;
        int hh  = h0 - 1 + row;                  // global input row
        int gc0 = w0 - 8 + run * 8;              // first global col of the run
        sRun[k] = run; sRow[k] = row; sCh[k] = ch;
        sOK[k]  = (hh >= 0) && (hh < H) && (gc0 >= 0) && (gc0 + 8 <= W);
        sOff[k] = (long)ch * chStride + (long)hh * W + gc0;   // add kb*chStride later
    }

    const long wTapStride = (long)Cout * Cin;

    for (int kb = 0; kb < Cin; kb += 32) {
        // -------- stage 32ch x 4x48 halo tile to LDS (once per K-slice) -----
        __syncthreads();
        #pragma unroll
        for (int k = 0; k < 3; ++k) {
            v8bf vals = {};
            if (sOK[k])                                   // uniform-ish, one 16B load
                vals = *(const v8bf*)(inB + (long)kb * chStride + sOff[k]);
            bf16* dst = &lds[(sRow[k] * LCOLS + sRun[k] * 8) * CSTR + sCh[k]];
            #pragma unroll
            for (int j = 0; j < 8; ++j) dst[j * CSTR] = vals[j];
        }
        __syncthreads();

        // prefetch next K-slice of weights (global_prefetch_b8), wave-uniform
        if (kb + 32 < Cin)
            __builtin_prefetch(wt + (long)coWave * Cin + kb + 32, 0, 3);

        // -------- 9 taps x 4 WMMAs against the staged window ---------------
        #pragma unroll
        for (int tap = 0; tap < 9; ++tap) {
            const int r = tap / 3 - 1, s = tap % 3 - 1;

            // A fragment: 16x32 bf16 weights (reused by 4 WMMAs)
            // lane<16: row=l16, K={kb+0..7, kb+16..23}; lane>=16: K shifted +8
            union { v16bf v; v8bf h[2]; } ua;
            {
                const bf16* wrow = wt + (long)tap * wTapStride
                                 + (long)(coWave + l16) * Cin + kb + halfSel * 8;
                ua.h[0] = *(const v8bf*)(wrow);
                ua.h[1] = *(const v8bf*)(wrow + 16);
            }

            #pragma unroll
            for (int q = 0; q < 4; ++q) {
                // B fragment: N=16 positions, K=32 channels from LDS
                // input row = h0 + (q>>1) + r -> tile row (q>>1)+r+1
                // input col = w0 + s + (q&1)*16 + l16 -> tile col 8+s+(q&1)*16+l16
                const int ri = (q >> 1) + r + 1;
                const int ci = 8 + s + (q & 1) * 16 + l16;
                union { v16bf v; v8bf h[2]; } ub;
                const bf16* p = &lds[(ri * LCOLS + ci) * CSTR + kOfs];
                ub.h[0] = *(const v8bf*)(p);
                ub.h[1] = *(const v8bf*)(p + 16);
                acc[q] = __builtin_amdgcn_wmma_f32_16x16x32_bf16(
                             false, ua.v, false, ub.v, (short)0, acc[q], false, false);
            }
        }
    }

    // epilogue: bias (+ReLU), store bf16. acc[q][j]: co = coWave + j + 8*halfSel,
    // row = h0 + (q>>1), col = w0 + (q&1)*16 + l16 -> 32B coalesced runs.
    #pragma unroll
    for (int j = 0; j < 8; ++j) {
        int coLoc = wave * 16 + j + halfSel * 8;
        float bv = bias[coBase + coLoc];
        bf16* q0 = out + ((long)b * Cout + coBase + coLoc) * H * W + (long)h0 * W + w0;
        float v0 = acc[0][j] + bv;
        float v1 = acc[1][j] + bv;
        float v2 = acc[2][j] + bv;
        float v3 = acc[3][j] + bv;
        if (relu) {
            v0 = fmaxf(v0, 0.f); v1 = fmaxf(v1, 0.f);
            v2 = fmaxf(v2, 0.f); v3 = fmaxf(v3, 0.f);
        }
        q0[l16]          = (bf16)v0;
        q0[l16 + 16]     = (bf16)v1;
        q0[W + l16]      = (bf16)v2;
        q0[W + l16 + 16] = (bf16)v3;
    }
}

// ---------------------------------------------------------------------------
// Final 3x3 conv layer, Cout = 1, f32 output (score map). M=1 makes WMMA
// wasteful; direct FMA loop is < 1e9 FLOPs total.
// ---------------------------------------------------------------------------
__global__ void k_conv3x3_last(const bf16* __restrict__ in, const bf16* __restrict__ wt,
                               const float* __restrict__ bias, float* __restrict__ out,
                               int Bn, int Cin, int H, int W)
{
    long idx = (long)blockIdx.x * blockDim.x + threadIdx.x;
    long total = (long)Bn * H * W;
    if (idx >= total) return;
    int w = (int)(idx % W); long t = idx / W;
    int h = (int)(t % H);   int b = (int)(t / H);
    float acc = bias[0];
    const bf16* inB = in + (long)b * Cin * H * W;
    for (int tap = 0; tap < 9; ++tap) {
        int r = tap / 3 - 1, s = tap % 3 - 1;
        int hh = h + r, ww = w + s;
        if (hh < 0 || hh >= H || ww < 0 || ww >= W) continue;
        const bf16* wp = wt + tap * Cin;       // [tap][1][Cin]
        const bf16* ip = inB + (long)hh * W + ww;
        for (int c = 0; c < Cin; ++c)
            acc += (float)ip[(long)c * H * W] * (float)wp[c];
    }
    out[idx] = acc;
}

// ---------------------------------------------------------------------------
// top-k (k=64) over N=9216 scores per batch row. One block per batch.
// Iterative extraction with block-wide argmax; ties -> lower index (matches
// jax.lax.top_k), output naturally sorted descending.
// ---------------------------------------------------------------------------
__global__ __launch_bounds__(256)
void k_topk64(const float* __restrict__ score, int* __restrict__ outIdx, int N, int K)
{
    __shared__ float sv[256];
    __shared__ int   si[256];
    const int b = blockIdx.x, tid = threadIdx.x;
    const float* s = score + (long)b * N;
    const float NEG = -__builtin_inff();
    float loc[36];                              // 9216 / 256
    #pragma unroll
    for (int i = 0; i < 36; ++i) {
        int p = tid + i * 256;
        loc[i] = (p < N) ? s[p] : NEG;
    }
    for (int k = 0; k < K; ++k) {
        float mv = NEG; int mi = 0x7fffffff;
        #pragma unroll
        for (int i = 0; i < 36; ++i) {
            int p = tid + i * 256;
            if (loc[i] > mv || (loc[i] == mv && p < mi)) { mv = loc[i]; mi = p; }
        }
        sv[tid] = mv; si[tid] = mi;
        __syncthreads();
        for (int off = 128; off > 0; off >>= 1) {
            if (tid < off) {
                float v2 = sv[tid + off]; int i2 = si[tid + off];
                if (v2 > sv[tid] || (v2 == sv[tid] && i2 < si[tid])) { sv[tid] = v2; si[tid] = i2; }
            }
            __syncthreads();
        }
        int win = si[0];
        if (tid == 0) outIdx[(long)b * K + k] = win;
        #pragma unroll
        for (int i = 0; i < 36; ++i)
            if (tid + i * 256 == win) loc[i] = NEG;
        __syncthreads();
    }
}

// ---------------------------------------------------------------------------
// Build match-head input grid bf16 [b][2C][K][K]:
//   c <  C : x[b][c][pick_idx[b][i]]   broadcast over j
//   c >= C : x[b][c-C][place_idx[b][j]] broadcast over i
// ---------------------------------------------------------------------------
__global__ void k_build_grid(const float* __restrict__ x,
                             const int* __restrict__ pick_idx,
                             const int* __restrict__ place_idx,
                             bf16* __restrict__ grid, int Bn, int C, int HW, int K)
{
    long idx = (long)blockIdx.x * blockDim.x + threadIdx.x;
    long total = (long)Bn * 2 * C * K * K;
    if (idx >= total) return;
    int j = (int)(idx % K); long t = idx / K;
    int i = (int)(t % K);   t /= K;
    int c = (int)(t % (2 * C));
    int b = (int)(t / (2 * C));
    int src_c, p;
    if (c < C) { src_c = c;     p = pick_idx[b * K + i]; }
    else       { src_c = c - C; p = place_idx[b * K + j]; }
    grid[idx] = (bf16)x[((long)b * C + src_c) * HW + p];
}

// ---------------------------------------------------------------------------
// Host side
// ---------------------------------------------------------------------------
static inline long ceil_div(long a, long b) { return (a + b - 1) / b; }

static void launch_conv(const bf16* in, const bf16* wt, const float* bias, bf16* out,
                        int Bn, int Cin, int Cout, int H, int W, int relu, hipStream_t st)
{
    dim3 grid((unsigned)(Bn * (H / 2) * (W / 32)), (unsigned)(Cout / 128));
    k_conv3x3_wmma<<<grid, 256, 0, st>>>(in, wt, bias, out, Cin, Cout, H, W, relu);
}

extern "C" void kernel_launch(void* const* d_in, const int* in_sizes, int n_in,
                              void* d_out, int out_size, void* d_ws, size_t ws_size,
                              hipStream_t stream)
{
    (void)in_sizes; (void)n_in; (void)out_size; (void)ws_size;
    constexpr int Bn = 8, C = 256, HID = 256, H = 96, W = 96, K = 64, HW = H * W;

    // inputs (leaf-flattened setup_inputs order)
    const float* x        = (const float*)d_in[0];
    const float* pick_w0  = (const float*)d_in[1];  const float* pick_b0  = (const float*)d_in[2];
    const float* pick_w1  = (const float*)d_in[3];  const float* pick_b1  = (const float*)d_in[4];
    const float* pick_w2  = (const float*)d_in[5];  const float* pick_b2  = (const float*)d_in[6];
    const float* place_w0 = (const float*)d_in[7];  const float* place_b0 = (const float*)d_in[8];
    const float* place_w1 = (const float*)d_in[9];  const float* place_b1 = (const float*)d_in[10];
    const float* place_w2 = (const float*)d_in[11]; const float* place_b2 = (const float*)d_in[12];
    const float* match_w0 = (const float*)d_in[13]; const float* match_b0 = (const float*)d_in[14];
    const float* match_w1 = (const float*)d_in[15]; const float* match_b1 = (const float*)d_in[16];
    const float* match_w2 = (const float*)d_in[17]; const float* match_b2 = (const float*)d_in[18];

    // d_out: match f32 (8*1*64*64) | pick_idx int (8*64) | place_idx int (8*64)
    float* match_out = (float*)d_out;
    int*   pick_idx  = (int*)d_out + (long)Bn * K * K;
    int*   place_idx = pick_idx + (long)Bn * K;

    // workspace carve-up (~188 MB)
    char* ws = (char*)d_ws;
    size_t off = 0;
    auto take = [&](size_t bytes) { char* p = ws + off; off = (off + bytes + 255) & ~(size_t)255; return p; };
    const long XN  = (long)Bn * C * HW;            // 18,874,368
    const long MGN = (long)Bn * 2 * C * K * K;     // 16,777,216
    const long MAN = (long)Bn * HID * K * K;       //  8,388,608
    bf16*  xb     = (bf16*)take(XN * 2);
    bf16*  act1   = (bf16*)take(XN * 2);
    bf16*  act2   = (bf16*)take(XN * 2);
    float* scoreP = (float*)take((long)Bn * HW * 4);
    float* scoreQ = (float*)take((long)Bn * HW * 4);
    bf16*  grid   = (bf16*)take(MGN * 2);
    bf16*  mact1  = (bf16*)take(MAN * 2);
    bf16*  mact2  = (bf16*)take(MAN * 2);
    bf16*  t_pw0  = (bf16*)take((long)HID * C * 9 * 2);
    bf16*  t_pw1  = (bf16*)take((long)HID * HID * 9 * 2);
    bf16*  t_pw2  = (bf16*)take((long)1 * HID * 9 * 2);
    bf16*  t_qw0  = (bf16*)take((long)HID * C * 9 * 2);
    bf16*  t_qw1  = (bf16*)take((long)HID * HID * 9 * 2);
    bf16*  t_qw2  = (bf16*)take((long)1 * HID * 9 * 2);
    bf16*  t_mw0  = (bf16*)take((long)HID * 2 * C * 9 * 2);
    bf16*  t_mw1  = (bf16*)take((long)HID * HID * 9 * 2);
    bf16*  t_mw2  = (bf16*)take((long)1 * HID * 9 * 2);

    // ---- one-time repacks ---------------------------------------------------
    k_f32_to_bf16<<<(unsigned)ceil_div(XN, 256), 256, 0, stream>>>(x, xb, XN);
    auto repack = [&](const float* w, bf16* o, int O, int I) {
        long n = (long)O * I * 9;
        k_repack_w<<<(unsigned)ceil_div(n, 256), 256, 0, stream>>>(w, o, O, I);
    };
    repack(pick_w0,  t_pw0, HID, C);   repack(pick_w1,  t_pw1, HID, HID);
    repack(pick_w2,  t_pw2, 1,   HID);
    repack(place_w0, t_qw0, HID, C);   repack(place_w1, t_qw1, HID, HID);
    repack(place_w2, t_qw2, 1,   HID);
    repack(match_w0, t_mw0, HID, 2*C); repack(match_w1, t_mw1, HID, HID);
    repack(match_w2, t_mw2, 1,   HID);

    // ---- pick head ----------------------------------------------------------
    launch_conv(xb,   t_pw0, pick_b0, act1, Bn, C,   HID, H, W, 1, stream);
    launch_conv(act1, t_pw1, pick_b1, act2, Bn, HID, HID, H, W, 1, stream);
    k_conv3x3_last<<<(unsigned)ceil_div((long)Bn * HW, 256), 256, 0, stream>>>(
        act2, t_pw2, pick_b2, scoreP, Bn, HID, H, W);
    k_topk64<<<Bn, 256, 0, stream>>>(scoreP, pick_idx, HW, K);

    // ---- place head ---------------------------------------------------------
    launch_conv(xb,   t_qw0, place_b0, act1, Bn, C,   HID, H, W, 1, stream);
    launch_conv(act1, t_qw1, place_b1, act2, Bn, HID, HID, H, W, 1, stream);
    k_conv3x3_last<<<(unsigned)ceil_div((long)Bn * HW, 256), 256, 0, stream>>>(
        act2, t_qw2, place_b2, scoreQ, Bn, HID, H, W);
    k_topk64<<<Bn, 256, 0, stream>>>(scoreQ, place_idx, HW, K);

    // ---- gather + outer-product grid ---------------------------------------
    k_build_grid<<<(unsigned)ceil_div(MGN, 256), 256, 0, stream>>>(
        x, pick_idx, place_idx, grid, Bn, C, HW, K);

    // ---- match head ---------------------------------------------------------
    launch_conv(grid,  t_mw0, match_b0, mact1, Bn, 2*C, HID, K, K, 1, stream);
    launch_conv(mact1, t_mw1, match_b1, mact2, Bn, HID, HID, K, K, 1, stream);
    k_conv3x3_last<<<(unsigned)ceil_div((long)Bn * K * K, 256), 256, 0, stream>>>(
        mact2, t_mw2, match_b2, match_out, Bn, HID, K, K);
}
